// Model_5583457485575
// MI455X (gfx1250) — matrix operand order repaired and checked by reading, not compiled, not run on hardware
//
#include <hip/hip_runtime.h>
#include <hip/hip_bf16.h>
#include <math.h>

// ---------------------------------------------------------------------------
// Model constants (from the reference)
// ---------------------------------------------------------------------------
#define BSZ   16
#define SEQL  512
#define HID   768
#define BSF   8
#define G3H   2304          // 3*HID
#define NROW  (BSZ*SEQL)    // 8192 GRU sequences / flattened rows
#define PAD_ID 1

typedef __attribute__((ext_vector_type(16))) _Float16 v16h;
typedef __attribute__((ext_vector_type(8)))  _Float16 v8h;
typedef __attribute__((ext_vector_type(8)))  float    v8f;
typedef __attribute__((ext_vector_type(4)))  float    v4f;

__device__ __forceinline__ v16h cat8h(v8h lo, v8h hi) {
  return __builtin_shufflevector(lo, hi, 0,1,2,3,4,5,6,7,8,9,10,11,12,13,14,15);
}

__device__ __forceinline__ float sigmoidf(float x) { return 1.f / (1.f + __expf(-x)); }

// ---------------------------------------------------------------------------
// WMMA GEMM:  C[M,N] = scale * (A[M,K] * B') + bias
//   A: f16 row-major [M][K]  (AMODE==0)  or rows gathered from an f32 table
//      (AMODE==1: row r -> tab + ids[r*idStride + idOff]*K, converted to f16)
//   B: f16 stored [N][K] row-major ("transposed" layout) -> per-lane fragment
//      is a contiguous 32B run of K.
//   Block = 256 threads (8 waves). Block tile 128(M) x 64(N); each wave does a
//   16x64 strip = 4 v_wmma_f32_16x16x32_f16 accumulators, K-loop step 32.
//   Optional f32 and/or f16 outputs; batched via blockIdx.z strides.
// ---------------------------------------------------------------------------
template<int AMODE>
__global__ __launch_bounds__(256)
void gemm_wmma_f16(const _Float16* __restrict__ A, int lda, long long sA,
                   const float* __restrict__ tab, const int* __restrict__ ids,
                   int idStride, int idOff,
                   const _Float16* __restrict__ B, int ldb, long long sB,
                   float* __restrict__ C, int ldc, long long sC,
                   _Float16* __restrict__ C16, int ldc16, long long sC16,
                   const float* __restrict__ bias, float scale,
                   int M, int N, int K)
{
  const int lane = threadIdx.x & 31;
  const int wave = threadIdx.x >> 5;
  const int bz   = blockIdx.z;
  const int m0   = blockIdx.y * 128 + wave * 16;
  const int n0   = blockIdx.x * 64;
  const int mrow = m0 + (lane & 15);
  const int kA   = (lane >> 4) * 16;   // A: lanes16-31 hold K+16 block
  const int kB   = (lane >> 4) * 16;   // B: lanes16-31 hold K+16 block

  const _Float16* Ap = nullptr;
  const float*    Af = nullptr;
  if (AMODE == 0) {
    Ap = A + (size_t)bz * sA + (size_t)mrow * lda + kA;
  } else {
    const int id = ids[(size_t)mrow * idStride + idOff];
    Af = tab + (size_t)id * lda + kA;
  }
  const _Float16* Bb = B + (size_t)bz * sB + kB;
  const _Float16* B0 = Bb + (size_t)(n0 + (lane & 15)) * ldb;
  const _Float16* B1 = B0 + (size_t)16 * ldb;
  const _Float16* B2 = B0 + (size_t)32 * ldb;
  const _Float16* B3 = B0 + (size_t)48 * ldb;

  v8f acc0 = {}, acc1 = {}, acc2 = {}, acc3 = {};

  for (int k = 0; k < K; k += 32) {
    // ---- A fragment (16 halves: K = kA..kA+15 within block)
    v16h a;
    if (AMODE == 0) {
      v8h lo = *(const v8h*)(Ap + k);
      v8h hi = *(const v8h*)(Ap + k + 8);
      a = cat8h(lo, hi);
    } else {
      v4f f0 = *(const v4f*)(Af + k);
      v4f f1 = *(const v4f*)(Af + k + 4);
      v4f f2 = *(const v4f*)(Af + k + 8);
      v4f f3 = *(const v4f*)(Af + k + 12);
      v8h lo, hi;
#pragma unroll
      for (int i = 0; i < 4; ++i) {
        lo[i]   = (_Float16)f0[i]; lo[4+i] = (_Float16)f1[i];
        hi[i]   = (_Float16)f2[i]; hi[4+i] = (_Float16)f3[i];
      }
      a = cat8h(lo, hi);
    }
    // ---- B fragments: contiguous 16-half runs
    v16h b0 = cat8h(*(const v8h*)(B0 + k), *(const v8h*)(B0 + k + 8));
    v16h b1 = cat8h(*(const v8h*)(B1 + k), *(const v8h*)(B1 + k + 8));
    v16h b2 = cat8h(*(const v8h*)(B2 + k), *(const v8h*)(B2 + k + 8));
    v16h b3 = cat8h(*(const v8h*)(B3 + k), *(const v8h*)(B3 + k + 8));

    acc0 = __builtin_amdgcn_wmma_f32_16x16x32_f16(false, a, false, b0, (short)0, acc0, false, false);
    acc1 = __builtin_amdgcn_wmma_f32_16x16x32_f16(false, a, false, b1, (short)0, acc1, false, false);
    acc2 = __builtin_amdgcn_wmma_f32_16x16x32_f16(false, a, false, b2, (short)0, acc2, false, false);
    acc3 = __builtin_amdgcn_wmma_f32_16x16x32_f16(false, a, false, b3, (short)0, acc3, false, false);
  }

  // ---- epilogue: C/D layout -> row = m0 + j + 8*(lane>>4), col = n0 + nt*16 + (lane&15)
  const int rbase = m0 + (lane >> 4) * 8;
  const int cbase = n0 + (lane & 15);
#pragma unroll
  for (int nt = 0; nt < 4; ++nt) {
    v8f av = (nt == 0) ? acc0 : (nt == 1) ? acc1 : (nt == 2) ? acc2 : acc3;
    const int col  = cbase + nt * 16;
    const float bv = bias ? bias[col] : 0.f;
#pragma unroll
    for (int j = 0; j < 8; ++j) {
      const int row = rbase + j;
      const float v = av[j] * scale + bv;
      if (C)   C  [(size_t)bz * sC   + (size_t)row * ldc   + col] = v;
      if (C16) C16[(size_t)bz * sC16 + (size_t)row * ldc16 + col] = (_Float16)v;
    }
  }
}

// ---------------------------------------------------------------------------
// Elementwise / helper kernels
// ---------------------------------------------------------------------------
__global__ void cvt_f32_f16_kernel(const float* __restrict__ s, _Float16* __restrict__ d, int n) {
  int i = blockIdx.x * 256 + threadIdx.x;
  if (i < n) d[i] = (_Float16)s[i];
}

// per-example DFG_index (= #pos>=2) and DFG_len (= #pos==0)
__global__ __launch_bounds__(256)
void pos_stats_kernel(const int* __restrict__ pos, int* __restrict__ dfg_index, int* __restrict__ dfg_len) {
  __shared__ int sT[256], sN[256];
  const int b = blockIdx.x;
  int t = 0, nn = 0;
  for (int j = threadIdx.x; j < SEQL; j += 256) {
    const int p = pos[b * SEQL + j];
    t  += (p >= 2);
    nn += (p == 0);
  }
  sT[threadIdx.x] = t; sN[threadIdx.x] = nn;
  __syncthreads();
  for (int s = 128; s > 0; s >>= 1) {
    if (threadIdx.x < s) { sT[threadIdx.x] += sT[threadIdx.x + s]; sN[threadIdx.x] += sN[threadIdx.x + s]; }
    __syncthreads();
  }
  if (threadIdx.x == 0) { dfg_index[b] = sT[0]; dfg_len[b] = sN[0]; }
}

// fin_DFG_ids[b][d][t] = (d + idx[b] < L) ? New_DFG_ids[b][d+idx[b]][t] : PAD
__global__ void fin_ids_kernel(const int* __restrict__ dfg_ids, const int* __restrict__ dfg_index,
                               int* __restrict__ fin) {
  const int idx = blockIdx.x * 256 + threadIdx.x;   // BSZ*SEQL*BSF
  if (idx >= BSZ * SEQL * BSF) return;
  const int t = idx & 7;
  const int d = (idx >> 3) & (SEQL - 1);
  const int b = idx >> 12;
  const int off = d + dfg_index[b];
  fin[idx] = (off < SEQL) ? dfg_ids[((b << 9) + off) * BSF + t] : PAD_ID;
}

// GRU gate fusion: r,z,n with PyTorch order; updates h (f32 + f16 copies)
__global__ void gru_gate_kernel(const float* __restrict__ Gx, const float* __restrict__ Gh,
                                const float* __restrict__ b_ih, const float* __restrict__ b_hh,
                                float* __restrict__ h32, _Float16* __restrict__ h16) {
  const int idx = blockIdx.x * 256 + threadIdx.x;   // NROW*HID
  const int n = idx / HID, c = idx - n * HID;
  const size_t g = (size_t)n * G3H + c;
  const float xr = Gx[g]            + b_ih[c];
  const float xz = Gx[g + HID]      + b_ih[HID + c];
  const float xn = Gx[g + 2 * HID]  + b_ih[2 * HID + c];
  const float hr = Gh[g]            + b_hh[c];
  const float hz = Gh[g + HID]      + b_hh[HID + c];
  const float hn = Gh[g + 2 * HID]  + b_hh[2 * HID + c];
  const float r  = sigmoidf(xr + hr);
  const float z  = sigmoidf(xz + hz);
  const float nn = tanhf(xn + r * hn);
  const float h  = h32[idx];
  const float hw = (1.f - z) * nn + z * h;
  h32[idx] = hw;
  h16[idx] = (_Float16)hw;
}

// DFGout = GRU last hidden + pos_emb[d]  (f16 for the attention GEMMs)
__global__ void dfgout_kernel(const float* __restrict__ h32, const float* __restrict__ pemb,
                              _Float16* __restrict__ dfg16) {
  const int idx = blockIdx.x * 256 + threadIdx.x;   // NROW*HID
  const int c = idx % HID;
  const int d = (idx / HID) & (SEQL - 1);
  dfg16[idx] = (_Float16)(h32[idx] + pemb[d * HID + c]);
}

// wave-per-row softmax over 512 cols; f16 probabilities out
__global__ __launch_bounds__(256)
void softmax_kernel(const float* __restrict__ S, _Float16* __restrict__ P) {
  const int row  = blockIdx.x * 8 + (threadIdx.x >> 5);
  const int lane = threadIdx.x & 31;
  const float* s = S + (size_t)row * SEQL;
  float x[16], mx = -1e30f;
#pragma unroll
  for (int i = 0; i < 16; ++i) { x[i] = s[lane + i * 32]; mx = fmaxf(mx, x[i]); }
#pragma unroll
  for (int o = 16; o; o >>= 1) mx = fmaxf(mx, __shfl_xor(mx, o, 32));
  float sum = 0.f;
#pragma unroll
  for (int i = 0; i < 16; ++i) { x[i] = __expf(x[i] - mx); sum += x[i]; }
#pragma unroll
  for (int o = 16; o; o >>= 1) sum += __shfl_xor(sum, o, 32);
  const float inv = 1.f / sum;
  _Float16* p = P + (size_t)row * SEQL;
#pragma unroll
  for (int i = 0; i < 16; ++i) p[lane + i * 32] = (_Float16)(x[i] * inv);
}

// A-mask (f16, 0/1) for the avg GEMM + per-row counts
__global__ __launch_bounds__(256)
void amask_kernel(const int* __restrict__ pos, const unsigned char* __restrict__ attn,
                  _Float16* __restrict__ amask, int* __restrict__ cnt) {
  __shared__ int red[256];
  const int i = blockIdx.x, b = blockIdx.y;
  const int node = (pos[b * SEQL + i] == 0);
  const size_t rowoff = ((size_t)(b * SEQL + i)) * SEQL;
  int local = 0;
  for (int j = threadIdx.x; j < SEQL; j += 256) {
    const int tok = (pos[b * SEQL + j] >= 2);
    const int am  = attn[rowoff + j] ? 1 : 0;
    const int m   = node & tok & am;
    amask[rowoff + j] = (_Float16)m;
    local += m;
  }
  red[threadIdx.x] = local;
  __syncthreads();
  for (int s = 128; s > 0; s >>= 1) {
    if (threadIdx.x < s) red[threadIdx.x] += red[threadIdx.x + s];
    __syncthreads();
  }
  if (threadIdx.x == 0) cnt[b * SEQL + i] = red[0];
}

// ET[b][h][j] = f16(word_emb[code[b][j]][h])   (gather + transpose via LDS tile)
__global__ __launch_bounds__(256)
void gather_transpose_ET(const int* __restrict__ code, const float* __restrict__ wemb,
                         _Float16* __restrict__ ET) {
  __shared__ _Float16 tile[16][17];
  const int tx = threadIdx.x & 15, ty = threadIdx.x >> 4;
  const int b = blockIdx.z;
  const int h = blockIdx.x * 16 + tx;
  const int j = blockIdx.y * 16 + ty;
  tile[ty][tx] = (_Float16)wemb[(size_t)code[b * SEQL + j] * HID + h];
  __syncthreads();
  const int ho = blockIdx.x * 16 + ty;
  const int jo = blockIdx.y * 16 + tx;
  ET[((size_t)b * HID + ho) * SEQL + jo] = tile[tx][ty];
}

// VT[b][h][j] = V[b][j][h]   (f16 transpose via LDS tile)
__global__ __launch_bounds__(256)
void transpose_f16_kernel(const _Float16* __restrict__ V, _Float16* __restrict__ VT) {
  __shared__ _Float16 tile[16][17];
  const int tx = threadIdx.x & 15, ty = threadIdx.x >> 4;
  const int b = blockIdx.z;
  const int h = blockIdx.x * 16 + tx;
  const int j = blockIdx.y * 16 + ty;
  tile[ty][tx] = V[((size_t)b * SEQL + j) * HID + h];
  __syncthreads();
  const int ho = blockIdx.x * 16 + ty;
  const int jo = blockIdx.y * 16 + tx;
  VT[((size_t)b * HID + ho) * SEQL + jo] = tile[tx][ty];
}

// final blend/select -> out (f32)
__global__ void final_kernel(const int* __restrict__ pos, const int* __restrict__ code,
                             const float* __restrict__ wemb,
                             const float* __restrict__ avg_raw, const int* __restrict__ cnt,
                             const float* __restrict__ ffout,
                             const int* __restrict__ dfg_index, const int* __restrict__ dfg_len,
                             float* __restrict__ out) {
  const int idx = blockIdx.x * 256 + threadIdx.x;   // BSZ*SEQL*HID
  const int c = idx % HID;
  const int i = (idx / HID) & (SEQL - 1);
  const int b = idx / (HID * SEQL);
  const int p = pos[b * SEQL + i];
  float val;
  if (p == 0) {
    const float denom = (float)cnt[b * SEQL + i] + 1e-10f;
    float a = avg_raw[idx] / denom;
    const int di = dfg_index[b], dl = dfg_len[b];
    if (i >= di && i < di + dl) {
      const float dv = ffout[(((size_t)b * SEQL) + (i - di)) * HID + c];
      a = 0.4f * a + 0.6f * dv;   // (1-ALPA)*avg + ALPA*dfg
    }
    val = a;
  } else {
    val = wemb[(size_t)code[b * SEQL + i] * HID + c];
  }
  out[idx] = val;
}

// ---------------------------------------------------------------------------
// Host orchestration
// ---------------------------------------------------------------------------
extern "C" void kernel_launch(void* const* d_in, const int* in_sizes, int n_in,
                              void* d_out, int out_size, void* d_ws, size_t ws_size,
                              hipStream_t stream) {
  const int*   code = (const int*)d_in[0];
  const unsigned char* attn = (const unsigned char*)d_in[1]; // jnp bool -> 1 byte
  const int*   pos  = (const int*)d_in[2];
  const int*   dfgi = (const int*)d_in[3];
  const float* wemb = (const float*)d_in[4];
  const float* pemb = (const float*)d_in[5];
  const float* Wq = (const float*)d_in[6];  const float* bq  = (const float*)d_in[7];
  const float* Wk = (const float*)d_in[8];  const float* bk  = (const float*)d_in[9];
  const float* Wv = (const float*)d_in[10]; const float* bv  = (const float*)d_in[11];
  const float* Wff= (const float*)d_in[12]; const float* bff = (const float*)d_in[13];
  const float* Wih= (const float*)d_in[14]; const float* Whh = (const float*)d_in[15];
  const float* bih= (const float*)d_in[16]; const float* bhh = (const float*)d_in[17];
  float* out = (float*)d_out;

  char* ws = (char*)d_ws;
  size_t off = 0;
  auto alloc = [&](size_t bytes) -> void* {
    void* p = ws + off;
    off += (bytes + 255) & ~(size_t)255;
    return p;
  };

  // ---- persistent workspace
  _Float16* Wih16 = (_Float16*)alloc((size_t)G3H * HID * 2);
  _Float16* Whh16 = (_Float16*)alloc((size_t)G3H * HID * 2);
  _Float16* Wq16  = (_Float16*)alloc((size_t)HID * HID * 2);
  _Float16* Wk16  = (_Float16*)alloc((size_t)HID * HID * 2);
  _Float16* Wv16  = (_Float16*)alloc((size_t)HID * HID * 2);
  _Float16* Wff16 = (_Float16*)alloc((size_t)HID * HID * 2);
  int*      fin   = (int*)alloc((size_t)BSZ * SEQL * BSF * 4);
  int*      didx  = (int*)alloc(BSZ * 4);
  int*      dlen  = (int*)alloc(BSZ * 4);
  int*      cnt   = (int*)alloc((size_t)NROW * 4);
  float*    h32   = (float*)alloc((size_t)NROW * HID * 4);
  _Float16* h16   = (_Float16*)alloc((size_t)NROW * HID * 2);
  _Float16* dfg16 = (_Float16*)alloc((size_t)NROW * HID * 2);
  float*    ffout = (float*)alloc((size_t)NROW * HID * 4);
  const size_t scratch0 = off;

  // ---- GRU scratch
  float* Gx = (float*)alloc((size_t)NROW * G3H * 4);
  float* Gh = (float*)alloc((size_t)NROW * G3H * 4);

  // ---- attention scratch reuses the GRU scratch region
  off = scratch0;
  _Float16* q16    = (_Float16*)alloc((size_t)NROW * HID * 2);
  _Float16* k16    = (_Float16*)alloc((size_t)NROW * HID * 2);
  _Float16* v16    = (_Float16*)alloc((size_t)NROW * HID * 2);
  _Float16* vT16   = (_Float16*)alloc((size_t)NROW * HID * 2);
  float*    scores = (float*)alloc((size_t)BSZ * SEQL * SEQL * 4);
  _Float16* p16    = (_Float16*)alloc((size_t)BSZ * SEQL * SEQL * 2);
  _Float16* ao16   = (_Float16*)alloc((size_t)NROW * HID * 2);
  _Float16* amask  = (_Float16*)alloc((size_t)BSZ * SEQL * SEQL * 2);
  _Float16* ET16   = (_Float16*)alloc((size_t)BSZ * HID * SEQL * 2);
  float*    avg    = (float*)alloc((size_t)NROW * HID * 4);
  (void)ws_size; (void)in_sizes; (void)n_in; (void)out_size;

  const int NB = 256;
  auto blks = [](long long n) { return (unsigned)((n + 255) / 256); };

  // 1) h0 = 0
  hipMemsetAsync(h32, 0, (size_t)NROW * HID * 4, stream);
  hipMemsetAsync(h16, 0, (size_t)NROW * HID * 2, stream);

  // 2) masks / indices
  pos_stats_kernel<<<BSZ, NB, 0, stream>>>(pos, didx, dlen);
  fin_ids_kernel<<<blks((long long)BSZ * SEQL * BSF), NB, 0, stream>>>(dfgi, didx, fin);

  // 3) f16 weight copies (layout stays [N][K] row-major -> WMMA B operand)
  cvt_f32_f16_kernel<<<blks((long long)G3H * HID), NB, 0, stream>>>(Wih, Wih16, G3H * HID);
  cvt_f32_f16_kernel<<<blks((long long)G3H * HID), NB, 0, stream>>>(Whh, Whh16, G3H * HID);
  cvt_f32_f16_kernel<<<blks((long long)HID * HID), NB, 0, stream>>>(Wq, Wq16, HID * HID);
  cvt_f32_f16_kernel<<<blks((long long)HID * HID), NB, 0, stream>>>(Wk, Wk16, HID * HID);
  cvt_f32_f16_kernel<<<blks((long long)HID * HID), NB, 0, stream>>>(Wv, Wv16, HID * HID);
  cvt_f32_f16_kernel<<<blks((long long)HID * HID), NB, 0, stream>>>(Wff, Wff16, HID * HID);

  // 4) GRU: 8 steps of (gathered x-GEMM, h-GEMM, fused gates)
  dim3 gGRU(G3H / 64, NROW / 128, 1);   // (36, 64)
  for (int t = 0; t < BSF; ++t) {
    gemm_wmma_f16<1><<<gGRU, NB, 0, stream>>>(
        nullptr, HID, 0, wemb, fin, BSF, t,
        Wih16, HID, 0,
        Gx, G3H, 0, nullptr, 0, 0,
        nullptr, 1.0f, NROW, G3H, HID);
    gemm_wmma_f16<0><<<gGRU, NB, 0, stream>>>(
        h16, HID, 0, nullptr, nullptr, 0, 0,
        Whh16, HID, 0,
        Gh, G3H, 0, nullptr, 0, 0,
        nullptr, 1.0f, NROW, G3H, HID);
    gru_gate_kernel<<<blks((long long)NROW * HID), NB, 0, stream>>>(Gx, Gh, bih, bhh, h32, h16);
  }

  // 5) DFGout = h + pos_emb (f16)
  dfgout_kernel<<<blks((long long)NROW * HID), NB, 0, stream>>>(h32, pemb, dfg16);

  // 6) q/k/v projections (f16 outputs only)
  dim3 gP(HID / 64, NROW / 128, 1);     // (12, 64)
  gemm_wmma_f16<0><<<gP, NB, 0, stream>>>(dfg16, HID, 0, nullptr, nullptr, 0, 0,
      Wq16, HID, 0, nullptr, 0, 0, q16, HID, 0, bq, 1.0f, NROW, HID, HID);
  gemm_wmma_f16<0><<<gP, NB, 0, stream>>>(dfg16, HID, 0, nullptr, nullptr, 0, 0,
      Wk16, HID, 0, nullptr, 0, 0, k16, HID, 0, bk, 1.0f, NROW, HID, HID);
  gemm_wmma_f16<0><<<gP, NB, 0, stream>>>(dfg16, HID, 0, nullptr, nullptr, 0, 0,
      Wv16, HID, 0, nullptr, 0, 0, v16, HID, 0, bv, 1.0f, NROW, HID, HID);

  // 7) v transpose for the probs@v GEMM B operand
  transpose_f16_kernel<<<dim3(HID / 16, SEQL / 16, BSZ), NB, 0, stream>>>(v16, vT16);

  // 8) scores = q k^T / sqrt(HID)   (batched; B = k in [N=j][K=h] layout)
  dim3 gS(SEQL / 64, SEQL / 128, BSZ);  // (8, 4, 16)
  gemm_wmma_f16<0><<<gS, NB, 0, stream>>>(
      q16, HID, (long long)SEQL * HID, nullptr, nullptr, 0, 0,
      k16, HID, (long long)SEQL * HID,
      scores, SEQL, (long long)SEQL * SEQL, nullptr, 0, 0,
      nullptr, 1.0f / sqrtf((float)HID), SEQL, SEQL, HID);

  // 9) softmax -> f16 probs
  softmax_kernel<<<NROW / 8, NB, 0, stream>>>(scores, p16);

  // 10) attn_out = probs @ v   (B = vT in [N=h][K=j] layout)
  dim3 gAV(HID / 64, SEQL / 128, BSZ);  // (12, 4, 16)
  gemm_wmma_f16<0><<<gAV, NB, 0, stream>>>(
      p16, SEQL, (long long)SEQL * SEQL, nullptr, nullptr, 0, 0,
      vT16, SEQL, (long long)HID * SEQL,
      nullptr, 0, 0, ao16, HID, (long long)SEQL * HID,
      nullptr, 1.0f, SEQL, HID, SEQL);

  // 11) DFGout_final = attn_out @ Wff^T + bff   (f32)
  gemm_wmma_f16<0><<<gP, NB, 0, stream>>>(ao16, HID, 0, nullptr, nullptr, 0, 0,
      Wff16, HID, 0, ffout, HID, 0, nullptr, 0, 0, bff, 1.0f, NROW, HID, HID);

  // 12) node/token/attn mask matrix + counts, token-embedding transpose
  amask_kernel<<<dim3(SEQL, BSZ), NB, 0, stream>>>(pos, attn, amask, cnt);
  gather_transpose_ET<<<dim3(HID / 16, SEQL / 16, BSZ), NB, 0, stream>>>(code, wemb, ET16);

  // 13) avg_raw = mask @ E   (normalized later, per-row, in f32)
  gemm_wmma_f16<0><<<gAV, NB, 0, stream>>>(
      amask, SEQL, (long long)SEQL * SEQL, nullptr, nullptr, 0, 0,
      ET16, SEQL, (long long)HID * SEQL,
      avg, HID, (long long)SEQL * HID, nullptr, 0, 0,
      nullptr, 1.0f, SEQL, HID, SEQL);

  // 14) blend + select -> out
  final_kernel<<<blks((long long)BSZ * SEQL * HID), NB, 0, stream>>>(
      pos, code, wemb, avg, cnt, ffout, didx, dlen, out);
}
